// MoERNN_29609504539069
// MI455X (gfx1250) — matrix operand order repaired and checked by reading, not compiled, not run on hardware
//
#include <hip/hip_runtime.h>
#include <hip/hip_bf16.h>

// ---------------------------------------------------------------------------
// MoE-BiLSTM for MI455X (gfx1250): bf16 WMMA everywhere a GEMM exists.
// Scan kernel register-blocks N 4-wide (8 WMMAs per operand-load batch) and
// keeps per-cell state off-register (LDS gate weights, L2-hot c-state) so the
// GEMM pipeline fits the 256-VGPR wave budget without scratch spills.
// Expert combine is race-free: per-expert bf16 slices summed in projection.
// ---------------------------------------------------------------------------

typedef __bf16 bf16_t;
typedef __attribute__((ext_vector_type(16))) __bf16 v16bf;
typedef __attribute__((ext_vector_type(8)))  __bf16 v8bf;
typedef __attribute__((ext_vector_type(8)))  float  v8f;

union V16 { v16bf v; v8bf h[2]; };

constexpr int B_   = 32;
constexpr int T_   = 512;
constexpr int I_   = 512;
constexpr int H_   = 512;
constexpr int G4H_ = 4 * H_;   // 2048
constexpr int E_   = 4;
constexpr int OUT_ = 512;

constexpr size_t NCOMB_E = (size_t)B_ * T_ * 2 * H_;   // per-expert combine slice

// LDS plan: gates fp32 [32][2048] + h bf16 [32][512] + gate weights [32]
constexpr size_t SCAN_GATES_BYTES = (size_t)B_ * G4H_ * sizeof(float);   // 256 KB
constexpr size_t SCAN_H_BYTES     = (size_t)B_ * H_ * sizeof(bf16_t);    //  32 KB
constexpr size_t SCAN_GE_BYTES    = (size_t)B_ * sizeof(float);          //  128 B
constexpr size_t SCAN_SMEM        = SCAN_GATES_BYTES + SCAN_H_BYTES + SCAN_GE_BYTES;

// ---------------------------------------------------------------------------
// Utility kernels
// ---------------------------------------------------------------------------
__global__ void cvt_bf16_kernel(const float* __restrict__ src,
                                bf16_t* __restrict__ dst, int n) {
    int i = blockIdx.x * blockDim.x + threadIdx.x;
    int stride = gridDim.x * blockDim.x;
    for (; i < n; i += stride) dst[i] = (bf16_t)src[i];
}

__global__ void addbias_kernel(const float* __restrict__ a,
                               const float* __restrict__ b,
                               float* __restrict__ c, int n) {
    int i = blockIdx.x * blockDim.x + threadIdx.x;
    if (i < n) c[i] = a[i] + b[i];
}

__global__ void zero_kernel(float* __restrict__ p, size_t n) {
    size_t i = (size_t)blockIdx.x * blockDim.x + threadIdx.x;
    size_t stride = (size_t)gridDim.x * blockDim.x;
    for (; i < n; i += stride) p[i] = 0.f;
}

// ---------------------------------------------------------------------------
// Router: mean over T, MLP(SiLU), softmax, top-2 gates + lb loss
// ---------------------------------------------------------------------------
__global__ void mean_kernel(const float* __restrict__ x, float* __restrict__ rin) {
    int o = blockIdx.x * blockDim.x + threadIdx.x;      // B*I = 16384
    if (o >= B_ * I_) return;
    int b = o >> 9, i = o & 511;
    const float* p = x + (size_t)b * T_ * I_ + i;
    float s = 0.f;
    for (int t = 0; t < T_; ++t) s += p[(size_t)t * I_];
    rin[o] = s * (1.0f / T_);
}

__global__ void router_h1_kernel(const float* __restrict__ rin,
                                 const float* __restrict__ rW1,
                                 const float* __restrict__ rb1,
                                 float* __restrict__ h1) {
    int o = blockIdx.x * blockDim.x + threadIdx.x;      // B*I
    if (o >= B_ * I_) return;
    int b = o >> 9, j = o & 511;
    const float* a = rin + b * I_;
    const float* w = rW1 + (size_t)j * I_;
    float s = rb1[j];
    for (int k = 0; k < I_; ++k) s += a[k] * w[k];
    h1[o] = s / (1.f + __expf(-s));                     // SiLU
}

__global__ __launch_bounds__(128)
void router_final_kernel(const float* __restrict__ h1,
                         const float* __restrict__ rW2,
                         const float* __restrict__ rb2,
                         float* __restrict__ gateOut,
                         float* __restrict__ lossOut) {
    __shared__ float logits[B_][E_];
    __shared__ float probs[B_][E_];
    int tid = threadIdx.x;                               // 128 = 32*4
    int b = tid >> 2, e = tid & 3;
    const float* a = h1 + b * I_;
    const float* w = rW2 + e * I_;
    float s = rb2[e];
    for (int k = 0; k < I_; ++k) s += a[k] * w[k];
    logits[b][e] = s;
    __syncthreads();
    if (tid < B_) {
        float l[4] = {logits[tid][0], logits[tid][1], logits[tid][2], logits[tid][3]};
        float mx = fmaxf(fmaxf(l[0], l[1]), fmaxf(l[2], l[3]));
        float p[4], sum = 0.f;
        for (int k = 0; k < 4; ++k) { p[k] = __expf(l[k] - mx); sum += p[k]; }
        for (int k = 0; k < 4; ++k) { p[k] /= sum; probs[tid][k] = p[k]; }
        int i1 = 0;
        for (int k = 1; k < 4; ++k) if (p[k] > p[i1]) i1 = k;
        int i2 = -1;
        for (int k = 0; k < 4; ++k) {
            if (k == i1) continue;
            if (i2 < 0 || p[k] > p[i2]) i2 = k;
        }
        float denom = p[i1] + p[i2];
        for (int k = 0; k < 4; ++k) {
            float g = (k == i1) ? p[i1] / denom : (k == i2 ? p[i2] / denom : 0.f);
            gateOut[tid * E_ + k] = g;
        }
    }
    __syncthreads();
    if (tid == 0) {
        float loss = 0.f;
        for (int e2 = 0; e2 < E_; ++e2) {
            float u = 0.f;
            for (int b2 = 0; b2 < B_; ++b2) u += probs[b2][e2];
            u *= (1.f / B_);
            float d = u - 1.f / E_;
            loss += d * d;
        }
        *lossOut = 0.01f * (loss / E_);
    }
}

// ---------------------------------------------------------------------------
// Persistent BiLSTM scan: one workgroup (16 waves) per (expert, direction).
// Wave w owns batch-tile mt = w&1 and 16 column-tiles processed as 4 groups
// of 4; per K-step one A pair (x_t,h) feeds 8 WMMAs on 4 accumulators.
// Cell c-state lives in an L2-hot global buffer; gate weights in LDS.
// Gate-weighted h is plain-stored (bf16) into this expert's combine slice.
// ---------------------------------------------------------------------------
__global__ __launch_bounds__(512)
void scan_kernel(const bf16_t* __restrict__ x_bf,    // [B,T,I]
                 const bf16_t* __restrict__ Wih_bf,  // [E,2,4H,I]
                 const bf16_t* __restrict__ Whh_bf,  // [E,2,4H,H]
                 const float*  __restrict__ bias,    // [E,2,4H] (bih+bhh)
                 const float*  __restrict__ gate,    // [B,E]
                 float* __restrict__ c_ws,           // [E*2,B,H] cell state
                 bf16_t* __restrict__ ecomb)         // [E,B,T,2H]
{
    extern __shared__ char smem[];
    float*  gatesLDS = (float*)smem;                        // [32][2048]
    bf16_t* hbuf  = (bf16_t*)(smem + SCAN_GATES_BYTES);     // [32][512]
    float*  geLDS = (float*)(smem + SCAN_GATES_BYTES + SCAN_H_BYTES); // [32]

    const int ed   = blockIdx.x;         // 0..7
    const int e    = ed >> 1;
    const int d    = ed & 1;
    const int tid  = threadIdx.x;
    const int wave = tid >> 5;
    const int lane = tid & 31;
    const bool hi  = lane >= 16;
    const int ml   = lane & 15;

    const bf16_t* Wih_e  = Wih_bf + (size_t)ed * G4H_ * I_;
    const bf16_t* Whh_e  = Whh_bf + (size_t)ed * G4H_ * H_;
    const float*  bias_e = bias + ed * G4H_;
    float*        c_blk  = c_ws + (size_t)ed * B_ * H_;
    bf16_t*       ec     = ecomb + (size_t)e * NCOMB_E + (size_t)d * H_;

    const int mt = wave & 1;             // fixed batch-tile per wave
    const int wg = wave >> 1;            // 0..7 column-group owner
    const int b0 = mt * 16;
    const int aoff = hi ? 8 : 0;
    const bf16_t* hr = hbuf + (b0 + ml) * H_;   // LDS A rows for h

    // init h = 0 in LDS; per-(b) gate weight for this expert into LDS
    for (int i = tid; i < B_ * H_; i += 512) hbuf[i] = (bf16_t)0.f;
    if (tid < B_) geLDS[tid] = gate[tid * E_ + e];
    __syncthreads();

    for (int t = 0; t < T_; ++t) {
        const int ta = d ? (T_ - 1 - t) : t;
        const bf16_t* xr = x_bf + ((size_t)(b0 + ml) * T_ + ta) * I_;

        // -------- GEMM phase: 4 groups x 4 column-tiles, K=512
        for (int g = 0; g < 4; ++g) {
            const int n0 = (wg * 4 + g * 32) * 16;   // column base of tile c=0
            const bf16_t* wi0 = Wih_e + (size_t)(n0 + ml) * I_ + (hi ? 8 : 0);
            const bf16_t* wh0 = Whh_e + (size_t)(n0 + ml) * H_ + (hi ? 8 : 0);

            v8f acc[4] = {{}, {}, {}, {}};
            for (int k0 = 0; k0 < I_; k0 += 32) {
                // A operands (shared by all 4 column tiles):
                // lo lanes K{0-7,16-23}, hi lanes K{8-15,24-31}
                V16 ax, ah;
                ax.h[0] = *(const v8bf*)(xr + k0 + aoff);
                ax.h[1] = *(const v8bf*)(xr + k0 + 16 + aoff);
                ah.h[0] = *(const v8bf*)(hr + k0 + aoff);
                ah.h[1] = *(const v8bf*)(hr + k0 + 16 + aoff);
                // B operands in distinct registers so loads batch in clauses
                V16 bi[4], bh[4];
#pragma unroll
                for (int c = 0; c < 4; ++c) {
                    const bf16_t* wiR = wi0 + (size_t)(c * 16) * I_ + k0;
                    const bf16_t* whR = wh0 + (size_t)(c * 16) * H_ + k0;
                    bi[c].h[0] = *(const v8bf*)(wiR);
                    bi[c].h[1] = *(const v8bf*)(wiR + 16);
                    bh[c].h[0] = *(const v8bf*)(whR);
                    bh[c].h[1] = *(const v8bf*)(whR + 16);
                }
                __builtin_prefetch(xr + k0 + 128, 0, 1);
                __builtin_prefetch(wi0 + k0 + 128, 0, 1);
                // 8 back-to-back WMMAs on independent accumulators
#pragma unroll
                for (int c = 0; c < 4; ++c) {
                    acc[c] = __builtin_amdgcn_wmma_f32_16x16x32_bf16(
                               false, ax.v, false, bi[c].v, (short)0, acc[c],
                               false, false);
                    acc[c] = __builtin_amdgcn_wmma_f32_16x16x32_bf16(
                               false, ah.v, false, bh[c].v, (short)0, acc[c],
                               false, false);
                }
            }
            const int rbase = b0 + (hi ? 8 : 0);
#pragma unroll
            for (int c = 0; c < 4; ++c) {
                const int ncol = n0 + c * 16 + ml;
                const float bn = bias_e[ncol];
#pragma unroll
                for (int r = 0; r < 8; ++r)
                    gatesLDS[(rbase + r) * G4H_ + ncol] = acc[c][r] + bn;
            }
        }
        __syncthreads();

        // -------- LSTM cell update: thread j owns hidden unit j for all b
        const int j = tid;   // 0..511
#pragma unroll 4
        for (int b = 0; b < B_; ++b) {
            float iv = gatesLDS[b * G4H_ + j];
            float fv = gatesLDS[b * G4H_ + H_ + j];
            float gv = gatesLDS[b * G4H_ + 2 * H_ + j];
            float ov = gatesLDS[b * G4H_ + 3 * H_ + j];
            iv = 1.f / (1.f + __expf(-iv));
            fv = 1.f / (1.f + __expf(-fv));
            gv = tanhf(gv);
            ov = 1.f / (1.f + __expf(-ov));
            float cOld = c_blk[b * H_ + j];
            float c = fv * cOld + iv * gv;
            c_blk[b * H_ + j] = c;
            float h = ov * tanhf(c);
            hbuf[b * H_ + j] = (bf16_t)h;
            // race-free, deterministic combine: this block owns (e, d) slice
            ec[((size_t)b * T_ + ta) * (2 * H_) + j] = (bf16_t)(geLDS[b] * h);
        }
        __syncthreads();
    }
}

// ---------------------------------------------------------------------------
// Output projection: A = sum_e ecomb[e] (bf16 loads, fp32 add, bf16 A) once
// per K-step, shared across 4 column accumulators; Wout bf16 B; + bout.
// ---------------------------------------------------------------------------
__global__ __launch_bounds__(256)
void proj_kernel(const bf16_t* __restrict__ ecomb,   // [E][B*T, 2H]
                 const bf16_t* __restrict__ Wout_bf, // [OUT, 2H]
                 const float*  __restrict__ bout,    // [OUT]
                 float* __restrict__ out)            // [B*T, OUT]
{
    const int tid  = threadIdx.x;
    const int wave = tid >> 5;
    const int lane = tid & 31;
    const bool hi  = lane >= 16;
    const int ml   = lane & 15;
    const int row0 = blockIdx.x * 16;                  // 1024 m-tiles
    const int K    = 2 * H_;                           // 1024
    const int n0   = wave * 64;                        // 4 tiles: n0 + c*16

    const bf16_t* ar0 = ecomb + (size_t)(row0 + ml) * K;   // expert-0 row
    const bf16_t* wr0 = Wout_bf + (size_t)(n0 + ml) * K + (hi ? 8 : 0);
    const int aoff = hi ? 8 : 0;

    v8f acc[4] = {{}, {}, {}, {}};
    for (int k0 = 0; k0 < K; k0 += 32) {
        V16 a;
#pragma unroll
        for (int ch = 0; ch < 2; ++ch) {
            const int off = k0 + ch * 16 + aoff;
            v8bf c0 = *(const v8bf*)(ar0 + off);
            v8bf c1 = *(const v8bf*)(ar0 + NCOMB_E + off);
            v8bf c2 = *(const v8bf*)(ar0 + 2 * NCOMB_E + off);
            v8bf c3 = *(const v8bf*)(ar0 + 3 * NCOMB_E + off);
#pragma unroll
            for (int z = 0; z < 8; ++z) {
                float s = (float)c0[z] + (float)c1[z] + (float)c2[z] + (float)c3[z];
                a.h[ch][z] = (bf16_t)s;
            }
        }
        V16 bm[4];
#pragma unroll
        for (int c = 0; c < 4; ++c) {
            const bf16_t* wr = wr0 + (size_t)(c * 16) * K + k0;
            bm[c].h[0] = *(const v8bf*)(wr);
            bm[c].h[1] = *(const v8bf*)(wr + 16);
        }
#pragma unroll
        for (int c = 0; c < 4; ++c)
            acc[c] = __builtin_amdgcn_wmma_f32_16x16x32_bf16(
                       false, a.v, false, bm[c].v, (short)0, acc[c],
                       false, false);
    }
    const int rbase = row0 + (hi ? 8 : 0);
#pragma unroll
    for (int c = 0; c < 4; ++c) {
        const int ncol = n0 + c * 16 + ml;
        const float bn = bout[ncol];
#pragma unroll
        for (int r = 0; r < 8; ++r)
            out[(size_t)(rbase + r) * OUT_ + ncol] = acc[c][r] + bn;
    }
}

// ---------------------------------------------------------------------------
// Host launch
// ---------------------------------------------------------------------------
extern "C" void kernel_launch(void* const* d_in, const int* in_sizes, int n_in,
                              void* d_out, int out_size, void* d_ws, size_t ws_size,
                              hipStream_t stream) {
    const float* x    = (const float*)d_in[0];
    const float* rW1  = (const float*)d_in[1];
    const float* rb1  = (const float*)d_in[2];
    const float* rW2  = (const float*)d_in[3];
    const float* rb2  = (const float*)d_in[4];
    const float* Wih  = (const float*)d_in[5];
    const float* Whh  = (const float*)d_in[6];
    const float* bih  = (const float*)d_in[7];
    const float* bhh  = (const float*)d_in[8];
    const float* Wout = (const float*)d_in[9];
    const float* bout = (const float*)d_in[10];
    float* out = (float*)d_out;   // [B*T*OUT] output + [1] lb_loss

    const size_t nX    = (size_t)B_ * T_ * I_;       // 8388608
    const size_t nWih  = (size_t)E_ * 2 * G4H_ * I_; // 8388608
    const size_t nWhh  = (size_t)E_ * 2 * G4H_ * H_; // 8388608
    const size_t nWout = (size_t)OUT_ * 2 * H_;      // 524288
    const size_t nBias = (size_t)E_ * 2 * G4H_;      // 16384
    const size_t nCst  = (size_t)E_ * 2 * B_ * H_;   // 131072
    const size_t nEC   = (size_t)E_ * NCOMB_E;       // 67108864 bf16

    char* w = (char*)d_ws;
    bf16_t* x_bf    = (bf16_t*)w; w += nX * sizeof(bf16_t);
    bf16_t* Wih_bf  = (bf16_t*)w; w += nWih * sizeof(bf16_t);
    bf16_t* Whh_bf  = (bf16_t*)w; w += nWhh * sizeof(bf16_t);
    bf16_t* Wout_bf = (bf16_t*)w; w += nWout * sizeof(bf16_t);
    float* bias_c   = (float*)w;  w += nBias * sizeof(float);
    float* rin      = (float*)w;  w += (size_t)B_ * I_ * sizeof(float);
    float* h1       = (float*)w;  w += (size_t)B_ * I_ * sizeof(float);
    float* gate     = (float*)w;  w += (size_t)B_ * E_ * sizeof(float);
    float* c_ws     = (float*)w;  w += nCst * sizeof(float);
    bf16_t* ecomb   = (bf16_t*)w; w += nEC * sizeof(bf16_t);

    // bf16 conversions + bias merge + zero cell state
    cvt_bf16_kernel<<<2048, 256, 0, stream>>>(x, x_bf, (int)nX);
    cvt_bf16_kernel<<<2048, 256, 0, stream>>>(Wih, Wih_bf, (int)nWih);
    cvt_bf16_kernel<<<2048, 256, 0, stream>>>(Whh, Whh_bf, (int)nWhh);
    cvt_bf16_kernel<<<512, 256, 0, stream>>>(Wout, Wout_bf, (int)nWout);
    addbias_kernel<<<64, 256, 0, stream>>>(bih, bhh, bias_c, (int)nBias);
    zero_kernel<<<64, 256, 0, stream>>>(c_ws, nCst);

    // router
    mean_kernel<<<64, 256, 0, stream>>>(x, rin);
    router_h1_kernel<<<64, 256, 0, stream>>>(rin, rW1, rb1, h1);
    router_final_kernel<<<1, 128, 0, stream>>>(h1, rW2, rb2, gate,
                                               out + (size_t)B_ * T_ * OUT_);

    // 8 persistent expert-direction scans, ~288 KB dynamic LDS each
    hipFuncSetAttribute((const void*)scan_kernel,
                        hipFuncAttributeMaxDynamicSharedMemorySize,
                        (int)SCAN_SMEM);
    scan_kernel<<<E_ * 2, 512, SCAN_SMEM, stream>>>(x_bf, Wih_bf, Whh_bf,
                                                    bias_c, gate, c_ws, ecomb);

    // output projection (sums expert slices on the fly)
    proj_kernel<<<(B_ * T_) / 16, 256, 0, stream>>>(ecomb, Wout_bf, bout, out);
}
